// SiglipEncoderLayer_34127810134495
// MI455X (gfx1250) — compile-verified, matrix operand order checked by reading, not run on hardware
//
#include <hip/hip_runtime.h>
#include <hip/hip_bf16.h>
#include <math.h>

// ---------------------------------------------------------------------------
// SigLIP encoder layer with 4:1 token merging, for gfx1250 (MI455X, wave32).
// All matmuls: v_wmma_f32_16x16x32_bf16. Tile staging uses
// global_load_async_to_lds_b128 (ASYNCcnt) when the builtin is available.
// ---------------------------------------------------------------------------

typedef __attribute__((ext_vector_type(16))) __bf16 v16bf;
typedef __attribute__((ext_vector_type(8)))  float  v8f;

union FragBF { v16bf v; uint4 u[2]; };

#define WG 256

// ---- async global->LDS copy (CDNA5), with safe fallback --------------------
#if defined(__has_builtin)
#  if __has_builtin(__builtin_amdgcn_global_load_async_to_lds_b128)
#    define HAVE_ASYNC_LDS 1
#  endif
#endif
#ifndef HAVE_ASYNC_LDS
#  define HAVE_ASYNC_LDS 0
#endif

// Builtin signature (from clang diagnostic): param0 is
// '__attribute__((vector_size(16))) int __device__ *'  i.e. v4i addrspace(1)*,
// param1 is the LDS-side v4i addrspace(3)*.
typedef __attribute__((__vector_size__(16))) int v4i_t;
typedef __attribute__((address_space(1))) v4i_t* gas_b128_ptr;
typedef __attribute__((address_space(3))) v4i_t* las_b128_ptr;

__device__ __forceinline__ void cp_b128_to_lds(__bf16* lds, const __bf16* g) {
#if HAVE_ASYNC_LDS
    // AS1 ptr is bit-identical to a generic global ptr (8B); AS3 ptr is the
    // low 32 bits of the generic LDS ptr.
    __builtin_amdgcn_global_load_async_to_lds_b128(
        __builtin_bit_cast(gas_b128_ptr, (const void*)g),
        __builtin_bit_cast(las_b128_ptr, (unsigned)(unsigned long long)lds),
        0, 0);
#else
    *(uint4*)lds = *(const uint4*)g;
#endif
}

__device__ __forceinline__ void wait_async_copies() {
#if HAVE_ASYNC_LDS
#  if defined(__has_builtin) && __has_builtin(__builtin_amdgcn_s_wait_asynccnt)
    __builtin_amdgcn_s_wait_asynccnt(0);
#  else
    asm volatile("s_wait_asynccnt 0x0" ::: "memory");
#  endif
#endif
}

// ---------------- fp32 -> bf16 conversion (weights) ----------------
__global__ __launch_bounds__(WG)
void f32_to_bf16_kernel(const float* __restrict__ in, __hip_bfloat16* __restrict__ out,
                        long long n) {
    long long i = (long long)blockIdx.x * WG + threadIdx.x;
    if (i < n) out[i] = __float2bfloat16(in[i]);
}

// ---------------- LayerNorm (per token) -> bf16 ----------------
__global__ __launch_bounds__(WG)
void ln_bf16_kernel(const float* __restrict__ x, const float* __restrict__ g,
                    const float* __restrict__ bet, __hip_bfloat16* __restrict__ out,
                    int Hdim) {
    __shared__ float r1[WG];
    __shared__ float r2[WG];
    const int t = threadIdx.x;
    const float* xr = x + (long long)blockIdx.x * Hdim;
    __hip_bfloat16* orow = out + (long long)blockIdx.x * Hdim;
    float s = 0.f, s2 = 0.f;
    for (int i = t; i < Hdim; i += WG) { float v = xr[i]; s += v; s2 += v * v; }
    r1[t] = s; r2[t] = s2; __syncthreads();
    for (int o = WG / 2; o > 0; o >>= 1) {
        if (t < o) { r1[t] += r1[t + o]; r2[t] += r2[t + o]; }
        __syncthreads();
    }
    const float mu  = r1[0] / (float)Hdim;
    const float var = r2[0] / (float)Hdim - mu * mu;
    const float inv = rsqrtf(var + 1e-6f);
    for (int i = t; i < Hdim; i += WG)
        orow[i] = __float2bfloat16((xr[i] - mu) * inv * g[i] + bet[i]);
}

// ---------------- merge K (row-major) and V (transposed) ----------------
__global__ __launch_bounds__(WG)
void merge_kernel(const __hip_bfloat16* __restrict__ Kin,
                  const __hip_bfloat16* __restrict__ Vin,
                  __hip_bfloat16* __restrict__ Km,
                  __hip_bfloat16* __restrict__ VmT) {
    const long long idx = (long long)blockIdx.x * WG + threadIdx.x;
    const long long total = 16LL * 256 * 1152;
    if (idx >= total) return;
    const int d = (int)(idx % 1152);
    const long long bj = idx / 1152;
    const int b = (int)(bj / 256);
    const int j = (int)(bj % 256);
    const long long base = ((long long)b * 1024 + (long long)j * 4) * 1152 + d;
    float ks = 0.f, vs = 0.f;
    for (int i = 0; i < 4; ++i) {
        ks += __bfloat162float(Kin[base + (long long)i * 1152]);
        vs += __bfloat162float(Vin[base + (long long)i * 1152]);
    }
    Km[idx] = __float2bfloat16(ks * 0.25f);
    VmT[((long long)b * 1152 + d) * 256 + j] = __float2bfloat16(vs * 0.25f);
}

// ---------------- softmax over 256 merged keys (scale folded in) ----------------
__global__ __launch_bounds__(WG)
void softmax_kernel(const float* __restrict__ S, __hip_bfloat16* __restrict__ P,
                    float scale) {
    __shared__ float red[WG];
    const int t = threadIdx.x;
    const long long row = (long long)blockIdx.x * 256;
    const float v = S[row + t] * scale;
    red[t] = v; __syncthreads();
    for (int o = WG / 2; o > 0; o >>= 1) {
        if (t < o) red[t] = fmaxf(red[t], red[t + o]);
        __syncthreads();
    }
    const float m = red[0]; __syncthreads();
    const float e = __expf(v - m);
    red[t] = e; __syncthreads();
    for (int o = WG / 2; o > 0; o >>= 1) {
        if (t < o) red[t] += red[t + o];
        __syncthreads();
    }
    P[row + t] = __float2bfloat16(e / red[0]);
}

// ---------------- WMMA GEMM: C = A[M x K] * W[N x K]^T (+epilogue) ----------------
// Block tile 128x64, K-step 32, 8 waves in 4(M) x 2(N), each wave -> 32x32 out
// (4 accumulators; 8 ds_load_b128 feed 4 WMMAs per K-step).
constexpr int EPI_BIAS_BF16      = 0;  // bf16 out = acc + bias          (Q/K/V)
constexpr int EPI_F32            = 1;  // f32  out = acc                 (scores)
constexpr int EPI_BF16           = 2;  // bf16 out = acc                 (attn*Vm)
constexpr int EPI_BIAS_RES_F32   = 3;  // f32  out = acc + bias + res    (Wo, W2)
constexpr int EPI_BIAS_GELU_BF16 = 4;  // bf16 out = gelu(acc + bias)    (W1)

template <int EPI>
__global__ __launch_bounds__(WG)
void wmma_gemm_kernel(const __bf16* __restrict__ Amat, long long sAb,
                      const __bf16* __restrict__ Wmat, long long sWb,
                      const float* __restrict__ bias,
                      const float* __restrict__ res,
                      void* __restrict__ Cout, long long sCb,
                      int Ncols, int Kdim) {
    constexpr int BM = 128, BN = 64, BK = 32, LDT = BK + 8;  // LDS stride 40 bf16
    __shared__ __bf16 As[BM * LDT];
    __shared__ __bf16 Bs[BN * LDT];

    const int zb = blockIdx.z;
    const __bf16* Ab = Amat + (long long)zb * sAb;
    const __bf16* Wb = Wmat + (long long)zb * sWb;

    const int tid  = threadIdx.x;
    const int lane = tid & 31;
    const int wave = tid >> 5;
    const int wm   = wave >> 1;          // 0..3 : 32-row slab
    const int wn   = wave & 1;           // 0..1 : 32-col slab
    const long long rowBase = (long long)blockIdx.x * BM;
    const long long colBase = (long long)blockIdx.y * BN;

    // staging: A has 512 b128 chunks (2/thread), B has 256 (1/thread)
    const int a0r = tid >> 2;                    // rows 0..63
    const int a1r = (tid + WG) >> 2;             // rows 64..127
    const int acg = (tid & 3) << 3;              // col offset 0,8,16,24
    const int br_ = tid >> 2;                    // rows 0..63
    const int bcg = acg;

    const int r16  = lane & 15;
    const int hsel = lane >> 4;          // 0: K {0..7,16..23}; 1: K {8..15,24..31}
    const int kofs = hsel << 3;

    v8f acc[2][2] = {};

    for (int k0 = 0; k0 < Kdim; k0 += BK) {
        cp_b128_to_lds(&As[a0r * LDT + acg],
                       Ab + (rowBase + a0r) * (long long)Kdim + k0 + acg);
        cp_b128_to_lds(&As[a1r * LDT + acg],
                       Ab + (rowBase + a1r) * (long long)Kdim + k0 + acg);
        cp_b128_to_lds(&Bs[br_ * LDT + bcg],
                       Wb + (colBase + br_) * (long long)Kdim + k0 + bcg);
        if (k0 + BK < Kdim) {  // global_prefetch_b8 hint for next K slab
            __builtin_prefetch(Ab + (rowBase + a0r) * (long long)Kdim + k0 + BK + acg, 0, 1);
            __builtin_prefetch(Wb + (colBase + br_) * (long long)Kdim + k0 + BK + bcg, 0, 1);
        }
        wait_async_copies();
        __syncthreads();

        FragBF a0, a1, b0, b1;
        const int ar = wm * 32 + r16;
        a0.u[0] = *(const uint4*)&As[ar * LDT + kofs];
        a0.u[1] = *(const uint4*)&As[ar * LDT + kofs + 16];
        a1.u[0] = *(const uint4*)&As[(ar + 16) * LDT + kofs];
        a1.u[1] = *(const uint4*)&As[(ar + 16) * LDT + kofs + 16];
        const int br2 = wn * 32 + r16;
        b0.u[0] = *(const uint4*)&Bs[br2 * LDT + kofs];
        b0.u[1] = *(const uint4*)&Bs[br2 * LDT + kofs + 16];
        b1.u[0] = *(const uint4*)&Bs[(br2 + 16) * LDT + kofs];
        b1.u[1] = *(const uint4*)&Bs[(br2 + 16) * LDT + kofs + 16];

        acc[0][0] = __builtin_amdgcn_wmma_f32_16x16x32_bf16(
            false, a0.v, false, b0.v, (short)0, acc[0][0], false, false);
        acc[0][1] = __builtin_amdgcn_wmma_f32_16x16x32_bf16(
            false, a0.v, false, b1.v, (short)0, acc[0][1], false, false);
        acc[1][0] = __builtin_amdgcn_wmma_f32_16x16x32_bf16(
            false, a1.v, false, b0.v, (short)0, acc[1][0], false, false);
        acc[1][1] = __builtin_amdgcn_wmma_f32_16x16x32_bf16(
            false, a1.v, false, b1.v, (short)0, acc[1][1], false, false);
        __syncthreads();
    }

    for (int n = 0; n < 2; ++n) {
        const long long col = colBase + wn * 32 + n * 16 + r16;
        float bval = 0.f;
        if constexpr (EPI == EPI_BIAS_BF16 || EPI == EPI_BIAS_RES_F32 ||
                      EPI == EPI_BIAS_GELU_BF16)
            bval = bias[col];
        for (int m = 0; m < 2; ++m) {
            for (int v = 0; v < 8; ++v) {
                const long long row  = rowBase + wm * 32 + m * 16 + hsel * 8 + v;
                const long long cidx = (long long)zb * sCb + row * Ncols + col;
                float val = acc[m][n][v] + bval;
                if constexpr (EPI == EPI_BIAS_BF16 || EPI == EPI_BF16) {
                    ((__hip_bfloat16*)Cout)[cidx] = __float2bfloat16(val);
                } else if constexpr (EPI == EPI_F32) {
                    ((float*)Cout)[cidx] = val;
                } else if constexpr (EPI == EPI_BIAS_RES_F32) {
                    ((float*)Cout)[cidx] = val + res[row * Ncols + col];
                } else {  // exact GELU
                    ((__hip_bfloat16*)Cout)[cidx] = __float2bfloat16(
                        0.5f * val * (1.0f + erff(val * 0.70710678118654752f)));
                }
            }
        }
    }
}

// ---------------------------------------------------------------------------
extern "C" void kernel_launch(void* const* d_in, const int* in_sizes, int n_in,
                              void* d_out, int out_size, void* d_ws, size_t ws_size,
                              hipStream_t stream) {
    const float* x    = (const float*)d_in[0];
    const float* ln1g = (const float*)d_in[1];
    const float* ln1b = (const float*)d_in[2];
    const float* ln2g = (const float*)d_in[3];
    const float* ln2b = (const float*)d_in[4];
    const float* Wq = (const float*)d_in[5];  const float* bq = (const float*)d_in[6];
    const float* Wk = (const float*)d_in[7];  const float* bk = (const float*)d_in[8];
    const float* Wv = (const float*)d_in[9];  const float* bv = (const float*)d_in[10];
    const float* Wo = (const float*)d_in[11]; const float* bo = (const float*)d_in[12];
    const float* W1 = (const float*)d_in[13]; const float* b1 = (const float*)d_in[14];
    const float* W2 = (const float*)d_in[15]; const float* b2 = (const float*)d_in[16];

    const long long Bb = 16, Nq = 1024, Hd = 1152, Md = 4608, Nm = 256;
    const long long T = Bb * Nq, Tm = Bb * Nm;

    // Workspace layout; M1 aliases [h|Q|K|V] which are all dead by then.
    char* ws = (char*)d_ws;
    __hip_bfloat16* hb  = (__hip_bfloat16*)(ws + 0 * T * Hd * 2);
    __hip_bfloat16* Qb  = (__hip_bfloat16*)(ws + 1 * T * Hd * 2);
    __hip_bfloat16* Kb  = (__hip_bfloat16*)(ws + 2 * T * Hd * 2);
    __hip_bfloat16* Vb  = (__hip_bfloat16*)(ws + 3 * T * Hd * 2);
    __hip_bfloat16* M1b = (__hip_bfloat16*)(ws);               // 16384 x 4608 bf16 alias
    size_t off = (size_t)(4 * T * Hd * 2);
    __hip_bfloat16* Kmb = (__hip_bfloat16*)(ws + off); off += (size_t)(Tm * Hd * 2);
    __hip_bfloat16* VmT = (__hip_bfloat16*)(ws + off); off += (size_t)(Tm * Hd * 2);
    float*          Sf  = (float*)         (ws + off); off += (size_t)(Bb * Nq * Nm * 4);
    __hip_bfloat16* Pb  = (__hip_bfloat16*)(ws + off); off += (size_t)(Bb * Nq * Nm * 2);
    __hip_bfloat16* Ob  = (__hip_bfloat16*)(ws + off); off += (size_t)(T * Hd * 2); // also h2
    float*          X1  = (float*)         (ws + off); off += (size_t)(T * Hd * 4);
    __hip_bfloat16* Wqb = (__hip_bfloat16*)(ws + off); off += (size_t)(Hd * Hd * 2);
    __hip_bfloat16* Wkb = (__hip_bfloat16*)(ws + off); off += (size_t)(Hd * Hd * 2);
    __hip_bfloat16* Wvb = (__hip_bfloat16*)(ws + off); off += (size_t)(Hd * Hd * 2);
    __hip_bfloat16* Wob = (__hip_bfloat16*)(ws + off); off += (size_t)(Hd * Hd * 2);
    __hip_bfloat16* W1b = (__hip_bfloat16*)(ws + off); off += (size_t)(Md * Hd * 2);
    __hip_bfloat16* W2b = (__hip_bfloat16*)(ws + off); off += (size_t)(Md * Hd * 2);

    const dim3 blk(WG);
    const unsigned nW  = (unsigned)((Hd * Hd + WG - 1) / WG);
    const unsigned nW2 = (unsigned)((Md * Hd + WG - 1) / WG);

    // 1) weights -> bf16
    f32_to_bf16_kernel<<<nW,  blk, 0, stream>>>(Wq, Wqb, Hd * Hd);
    f32_to_bf16_kernel<<<nW,  blk, 0, stream>>>(Wk, Wkb, Hd * Hd);
    f32_to_bf16_kernel<<<nW,  blk, 0, stream>>>(Wv, Wvb, Hd * Hd);
    f32_to_bf16_kernel<<<nW,  blk, 0, stream>>>(Wo, Wob, Hd * Hd);
    f32_to_bf16_kernel<<<nW2, blk, 0, stream>>>(W1, W1b, Md * Hd);
    f32_to_bf16_kernel<<<nW2, blk, 0, stream>>>(W2, W2b, Md * Hd);

    // 2) LN1
    ln_bf16_kernel<<<(unsigned)T, blk, 0, stream>>>(x, ln1g, ln1b, hb, (int)Hd);

    // 3) QKV projections
    const dim3 gQKV((unsigned)(T / 128), (unsigned)(Hd / 64), 1);
    wmma_gemm_kernel<EPI_BIAS_BF16><<<gQKV, blk, 0, stream>>>(
        (const __bf16*)hb, 0, (const __bf16*)Wqb, 0, bq, nullptr, Qb, 0, (int)Hd, (int)Hd);
    wmma_gemm_kernel<EPI_BIAS_BF16><<<gQKV, blk, 0, stream>>>(
        (const __bf16*)hb, 0, (const __bf16*)Wkb, 0, bk, nullptr, Kb, 0, (int)Hd, (int)Hd);
    wmma_gemm_kernel<EPI_BIAS_BF16><<<gQKV, blk, 0, stream>>>(
        (const __bf16*)hb, 0, (const __bf16*)Wvb, 0, bv, nullptr, Vb, 0, (int)Hd, (int)Hd);

    // 4) token merge (r = 4); Vm stored transposed
    merge_kernel<<<(unsigned)((Tm * Hd + WG - 1) / WG), blk, 0, stream>>>(Kb, Vb, Kmb, VmT);

    // 5) scores = Q @ Km^T  (log(r) cancels in softmax; scale folded into softmax)
    const dim3 gS((unsigned)(Nq / 128), (unsigned)(Nm / 64), (unsigned)Bb);
    wmma_gemm_kernel<EPI_F32><<<gS, blk, 0, stream>>>(
        (const __bf16*)Qb, Nq * Hd, (const __bf16*)Kmb, Nm * Hd, nullptr, nullptr,
        Sf, Nq * Nm, (int)Nm, (int)Hd);

    // 6) softmax
    softmax_kernel<<<(unsigned)T, blk, 0, stream>>>(Sf, Pb, 1.0f / sqrtf((float)Hd));

    // 7) o = attn @ Vm   (B operand is VmT[b] : [1152 x 256] row-major)
    const dim3 gAV((unsigned)(Nq / 128), (unsigned)(Hd / 64), (unsigned)Bb);
    wmma_gemm_kernel<EPI_BF16><<<gAV, blk, 0, stream>>>(
        (const __bf16*)Pb, Nq * Nm, (const __bf16*)VmT, Hd * Nm, nullptr, nullptr,
        Ob, Nq * Hd, (int)Hd, (int)Nm);

    // 8) x1 = x + o @ Wo^T + bo
    const dim3 gO((unsigned)(T / 128), (unsigned)(Hd / 64), 1);
    wmma_gemm_kernel<EPI_BIAS_RES_F32><<<gO, blk, 0, stream>>>(
        (const __bf16*)Ob, 0, (const __bf16*)Wob, 0, bo, x, X1, 0, (int)Hd, (int)Hd);

    // 9) LN2 (h2 reuses Ob's buffer — Wo GEMM already consumed it)
    __hip_bfloat16* H2b = Ob;
    ln_bf16_kernel<<<(unsigned)T, blk, 0, stream>>>(X1, ln2g, ln2b, H2b, (int)Hd);

    // 10) m1 = gelu(h2 @ W1^T + b1)
    const dim3 g1((unsigned)(T / 128), (unsigned)(Md / 64), 1);
    wmma_gemm_kernel<EPI_BIAS_GELU_BF16><<<g1, blk, 0, stream>>>(
        (const __bf16*)H2b, 0, (const __bf16*)W1b, 0, b1, nullptr, M1b, 0, (int)Md, (int)Hd);

    // 11) out = x1 + m1 @ W2^T + b2
    const dim3 g2((unsigned)(T / 128), (unsigned)(Hd / 64), 1);
    wmma_gemm_kernel<EPI_BIAS_RES_F32><<<g2, blk, 0, stream>>>(
        (const __bf16*)M1b, 0, (const __bf16*)W2b, 0, b2, X1, (float*)d_out, 0,
        (int)Hd, (int)Md);
}
